// Decoder_18348100289269
// MI455X (gfx1250) — matrix operand order repaired and checked by reading, not compiled, not run on hardware
//
#include <hip/hip_runtime.h>

typedef __attribute__((ext_vector_type(16))) __bf16 v16bf;
typedef __attribute__((ext_vector_type(8)))  __bf16 v8bf;
typedef __attribute__((ext_vector_type(8)))  float  v8f;
typedef __attribute__((ext_vector_type(4)))  unsigned int v4u;
typedef __attribute__((ext_vector_type(8)))  int v8i;
typedef __attribute__((ext_vector_type(4)))  int v4i;

namespace {
constexpr int Dm = 768;          // embed dim
constexpr int Sq = 1024;         // seq len
constexpr int Bn = 4;            // batch
constexpr int Hh = 12;           // heads
constexpr int HD = 64;           // head dim
constexpr int Mrows = Bn * Sq;   // 4096 tokens
}

// ---------- helpers ----------
static __device__ __forceinline__ v16bf ldpair(const __bf16* p0, const __bf16* p1) {
  v8bf a = *(const v8bf*)p0;
  v8bf b = *(const v8bf*)p1;
  v16bf r;
#pragma unroll
  for (int i = 0; i < 8; ++i) { r[i] = a[i]; r[i + 8] = b[i]; }
  return r;
}
// B-fragment: 16 contiguous K values for this lane-half
static __device__ __forceinline__ v16bf ldB16(const __bf16* p) { return ldpair(p, p + 8); }
// A-fragment: 16 contiguous K values {b0..b0+15}
static __device__ __forceinline__ v16bf ldA16(const __bf16* p) { return ldpair(p, p + 8); }

// ---------- Tensor Data Mover: load a 2D bf16 tile [tile1 rows x 32 cols] ----------
// Tensor is row-major [768 x K] bf16 (K = row stride, elements). Tile rows packed
// contiguously into LDS at ldsOff (64 bytes per row).
static __device__ __forceinline__ void tdm_load_w_tile(const __bf16* gaddr,
                                                       unsigned ldsOff, int K) {
  unsigned long long ga = (unsigned long long)(const void*)gaddr;
  v4u g0;
  g0[0] = 1u;                                            // count=1 (valid user D#)
  g0[1] = ldsOff;                                        // lds_addr (bytes)
  g0[2] = (unsigned)(ga & 0xffffffffu);                  // global_addr[31:0]
  g0[3] = (unsigned)((ga >> 32) & 0x1ffffffu) | (2u << 30); // addr[56:32] | type=2
  v8i g1;
  g1[0] = 1 << 16;                                       // wg_mask=0, data_size=1 (2B)
  g1[1] = (int)((unsigned)(K & 0xffff) << 16);           // tensor_dim0[15:0] = K
  g1[2] = (int)(768u << 16);                             // tensor_dim1[15:0] = 768
  g1[3] = (int)(32u << 16);                              // tile_dim0 = 32
  g1[4] = 64;                                            // tile_dim1 = 64, tile_dim2 = 0
  g1[5] = K;                                             // tensor_dim0_stride[31:0] = K
  g1[6] = 0;
  g1[7] = 0;
  v4i z4 = {0, 0, 0, 0};
#if defined(__clang_major__) && __clang_major__ >= 23
  v8i z8 = {0, 0, 0, 0, 0, 0, 0, 0};
  __builtin_amdgcn_tensor_load_to_lds(g0, g1, z4, z4, z8, 0);
#else
  __builtin_amdgcn_tensor_load_to_lds(g0, g1, z4, z4, 0);
#endif
}

// ---------- one-time converts ----------
__global__ void k_convT(const float* __restrict__ w, __bf16* __restrict__ wt, int K, int N) {
  int i = blockIdx.x * blockDim.x + threadIdx.x;
  if (i >= K * N) return;
  int k = i / N, n = i % N;
  wt[(size_t)n * K + k] = (__bf16)w[i];          // store W^T [N,K] in bf16
}

__global__ void k_conv(const float* __restrict__ s, __bf16* __restrict__ d, int n) {
  int i = blockIdx.x * blockDim.x + threadIdx.x;
  if (i < n) d[i] = (__bf16)s[i];
}

// ---------- embedding + positional encoding: out = 2*embed[x] + pos ----------
__global__ void k_embed(const int* __restrict__ x, const float* __restrict__ emb,
                        float* __restrict__ outF, __bf16* __restrict__ outB) {
  int i = blockIdx.x * blockDim.x + threadIdx.x;
  if (i >= Mrows * Dm) return;
  int d = i % Dm;
  int t = i / Dm;
  int s = t % Sq;
  int tok = x[t];
  float fi = (float)(d >> 1);
  float ang = (float)s * __expf(fi * (-2.0f / (float)Dm) * 9.210340371976184f); // /10000^(2i/D)
  float pos = (d & 1) ? __cosf(ang) : __sinf(ang);
  float v = 2.0f * emb[(size_t)tok * Dm + d] + pos;
  outF[i] = v;
  outB[i] = (__bf16)v;
}

// ---------- WMMA GEMM with TDM-staged weights ----------
// out[4096,768] = act(A[4096,K](bf16,lda) @ W + bias); Wt = W^T [768,K] bf16.
// Block = 8 waves sharing one 64-col weight slice (n0 block-uniform); wave ->
// 16(M) x 64(N) tile. Weight tiles (64x32 bf16, 4KB) are streamed into LDS by the
// Tensor Data Mover (double-buffered, TENSORcnt-tracked); WMMAs read B-fragments
// from LDS, A-fragments from global.
__global__ __launch_bounds__(256)
void k_gemm(const __bf16* __restrict__ A, int lda,
            const __bf16* __restrict__ Wt, int K,
            const float* __restrict__ bias, int doRelu,
            float* __restrict__ outF, __bf16* __restrict__ outB,
            __bf16* __restrict__ outBT /* [B,H,64,S] per-head transposed, or null */) {
  __shared__ __align__(16) __bf16 sW[2][64 * 32];   // two 4KB weight-tile buffers
  const int lane = threadIdx.x & 31;
  const int wid  = threadIdx.x >> 5;
  const int tile = blockIdx.x * 8 + wid;       // 3072 tiles total
  const int row0 = (tile & 255) * 16;          // M/16 = 256
  const int n0   = (tile >> 8) * 64;           // N/64 = 12 (uniform across block)
  const int hf   = lane >> 4;                  // lane half
  const int mr   = lane & 15;
  const int b0   = hf ? 16 : 0;                // A-frag K base

  v8f acc[4] = {v8f{}, v8f{}, v8f{}, v8f{}};
  const __bf16* arow = A + (size_t)(row0 + mr) * lda;
  const unsigned ldsBase = (unsigned)(unsigned long long)(void*)&sW[0][0];
  const int nsteps = K >> 5;

  if (wid == 0) tdm_load_w_tile(Wt + (size_t)n0 * K, ldsBase, K);

  for (int i = 0; i < nsteps; ++i) {
    const int k0 = i << 5;
    if (wid == 0) __builtin_amdgcn_s_wait_tensorcnt(0);   // current buffer landed
    __syncthreads();                                      // publish LDS to all waves
    if (i + 1 < nsteps && wid == 0)                       // overlap next DMA w/ math
      tdm_load_w_tile(Wt + (size_t)n0 * K + (k0 + 32),
                      ldsBase + (unsigned)(((i + 1) & 1) * 4096), K);
    __builtin_prefetch(arow + k0 + 128, 0, 3);            // near-cache A prefetch
    const __bf16* wtile = &sW[i & 1][0];
    v16bf bf4[4];
#pragma unroll
    for (int t = 0; t < 4; ++t)
      bf4[t] = ldB16(wtile + (t * 16 + mr) * 32 + hf * 16);
    v16bf af = ldA16(arow + k0 + b0);
#pragma unroll
    for (int t = 0; t < 4; ++t)
      acc[t] = __builtin_amdgcn_wmma_f32_16x16x32_bf16(false, af, false, bf4[t],
                                                       (short)0, acc[t], false, false);
    __syncthreads();                                      // done reading this buffer
  }

#pragma unroll
  for (int t = 0; t < 4; ++t) {
    const int col = n0 + t * 16 + mr;
    const float bv = bias[col];
#pragma unroll
    for (int v = 0; v < 8; ++v) {
      const int row = row0 + v + 8 * hf;
      float val = acc[t][v] + bv;
      if (doRelu) val = fmaxf(val, 0.0f);
      const size_t idx = (size_t)row * Dm + col;
      if (outF) outF[idx] = val;
      if (outB) outB[idx] = (__bf16)val;
      if (outBT) {
        int bb = row / Sq, ss = row % Sq;
        int hh = col / HD, dd = col % HD;
        outBT[(((size_t)(bb * Hh + hh) * HD + dd) * Sq) + ss] = (__bf16)val;
      }
    }
  }
}

// ---------- fused flash attention ----------
// Q,K: [B,S,768] bf16 (head slice = col h*64). Vt: [B,H,64,S] bf16.
// One wave per (b,h,16-row q tile). 384 blocks x 8 waves.
__global__ __launch_bounds__(256)
void k_flash(const __bf16* __restrict__ Q, const __bf16* __restrict__ Kb,
             const __bf16* __restrict__ Vt, float scale,
             float* __restrict__ outF, __bf16* __restrict__ outB) {
  __shared__ __align__(16) float          sS[8][16 * 32];
  __shared__ __align__(16) unsigned short sP[8][16 * 32];
  const int lane = threadIdx.x & 31;
  const int wid  = threadIdx.x >> 5;
  const int job  = blockIdx.x * 8 + wid;        // 3072 jobs
  const int q0   = (job & 63) * 16;
  const int h    = (job >> 6) % Hh;
  const int b    = job / (64 * Hh);
  const int hf   = lane >> 4;
  const int mr   = lane & 15;
  const int b0   = hf ? 16 : 0;

  const __bf16* qrow = Q + ((size_t)(b * Sq) + q0 + mr) * Dm + h * HD;
  v16bf qf0 = ldA16(qrow + b0);        // d 0..31
  v16bf qf1 = ldA16(qrow + 32 + b0);   // d 32..63

  float rmax = -1e30f, rsum = 0.0f;    // lanes 0..15 own row mr
  v8f acc[4] = {v8f{}, v8f{}, v8f{}, v8f{}};
  float*          myS = sS[wid];
  unsigned short* myP = sP[wid];

  for (int key0 = 0; key0 < Sq; key0 += 32) {
    // ---- scores 16q x 32k ----
#pragma unroll
    for (int c = 0; c < 2; ++c) {
      const __bf16* krow = Kb + ((size_t)(b * Sq) + key0 + c * 16 + mr) * Dm + h * HD;
      v16bf kf0 = ldB16(krow + hf * 16);        // d 0..31 (B-frag)
      v16bf kf1 = ldB16(krow + 32 + hf * 16);   // d 32..63
      v8f s = {};
      s = __builtin_amdgcn_wmma_f32_16x16x32_bf16(false, qf0, false, kf0, (short)0, s, false, false);
      s = __builtin_amdgcn_wmma_f32_16x16x32_bf16(false, qf1, false, kf1, (short)0, s, false, false);
#pragma unroll
      for (int v = 0; v < 8; ++v)
        myS[(v + 8 * hf) * 32 + c * 16 + mr] = s[v] * scale;
    }
    __syncthreads();
    // ---- online softmax: lanes 0..15, row = mr ----
    float corr = 0.0f;
    if (lane < 16) {
      float m = rmax;
#pragma unroll 4
      for (int j = 0; j < 32; ++j) m = fmaxf(m, myS[mr * 32 + j]);
      corr = __expf(rmax - m);
      float sum = rsum * corr;
#pragma unroll 4
      for (int j = 0; j < 32; ++j) {
        float p = __expf(myS[mr * 32 + j] - m);
        sum += p;
        myP[mr * 32 + j] = __builtin_bit_cast(unsigned short, (__bf16)p);
      }
      rmax = m; rsum = sum;
    }
    __syncthreads();
    // ---- rescale accumulators ----
#pragma unroll
    for (int v = 0; v < 8; ++v) {
      float cr = __shfl(corr, v + 8 * hf, 32);
#pragma unroll
      for (int t = 0; t < 4; ++t) acc[t][v] *= cr;
    }
    // ---- P (16x32, A-frag from LDS) @ V (32x64) ----
    v16bf pf = ldA16((const __bf16*)(myP + mr * 32 + b0));
#pragma unroll
    for (int t = 0; t < 4; ++t) {
      const __bf16* vrow = Vt + (((size_t)(b * Hh + h) * HD) + t * 16 + mr) * Sq + key0 + hf * 16;
      v16bf vf = ldB16(vrow);
      acc[t] = __builtin_amdgcn_wmma_f32_16x16x32_bf16(false, pf, false, vf,
                                                       (short)0, acc[t], false, false);
    }
  }
  // ---- epilogue: divide by row sum, write [B,S,768] ----
  float inv = (lane < 16) ? 1.0f / rsum : 0.0f;
#pragma unroll
  for (int v = 0; v < 8; ++v) {
    float f = __shfl(inv, v + 8 * hf, 32);
    const int qq = q0 + v + 8 * hf;
    const size_t base = ((size_t)(b * Sq) + qq) * Dm + h * HD;
#pragma unroll
    for (int t = 0; t < 4; ++t) {
      float val = acc[t][v] * f;
      outF[base + t * 16 + mr] = val;
      outB[base + t * 16 + mr] = (__bf16)val;
    }
  }
}

// ---------- batch-norm (training stats over B*S per feature) ----------
__global__ __launch_bounds__(256)
void k_bnstats(const float* __restrict__ X, const float* __restrict__ Y, float* __restrict__ st) {
  __shared__ float s1[256], s2[256];
  const int d = blockIdx.x;
  float a = 0.0f, b = 0.0f;
  for (int r = threadIdx.x; r < Mrows; r += 256) {
    float v = X[(size_t)r * Dm + d] + Y[(size_t)r * Dm + d];
    a += v; b += v * v;
  }
  s1[threadIdx.x] = a; s2[threadIdx.x] = b;
  __syncthreads();
  for (int off = 128; off; off >>= 1) {
    if (threadIdx.x < off) {
      s1[threadIdx.x] += s1[threadIdx.x + off];
      s2[threadIdx.x] += s2[threadIdx.x + off];
    }
    __syncthreads();
  }
  if (threadIdx.x == 0) {
    float mu  = s1[0] / (float)Mrows;
    float var = s2[0] / (float)Mrows - mu * mu;
    st[d]      = mu;
    st[Dm + d] = rsqrtf(var + 1e-5f);
  }
}

__global__ void k_bnapply(const float* __restrict__ X, const float* __restrict__ Y,
                          const float* __restrict__ st, const float* __restrict__ g,
                          const float* __restrict__ bb,
                          float* __restrict__ outF, __bf16* __restrict__ outB) {
  int i = blockIdx.x * blockDim.x + threadIdx.x;
  if (i >= Mrows * Dm) return;
  int d = i % Dm;
  float v = (X[i] + Y[i] - st[d]) * st[Dm + d] * g[d] + bb[d];
  outF[i] = v;
  outB[i] = (__bf16)v;
}

// ---------- host ----------
extern "C" void kernel_launch(void* const* d_in, const int* in_sizes, int n_in,
                              void* d_out, int out_size, void* d_ws, size_t ws_size,
                              hipStream_t stream) {
  (void)in_sizes; (void)n_in; (void)out_size; (void)ws_size;
  const int*   x     = (const int*)d_in[0];
  const float* encod = (const float*)d_in[1];
  const float* embed = (const float*)d_in[2];
  const float* Wq  = (const float*)d_in[3];  const float* bq  = (const float*)d_in[4];
  const float* Wk  = (const float*)d_in[5];  const float* bk  = (const float*)d_in[6];
  const float* Wv  = (const float*)d_in[7];  const float* bv  = (const float*)d_in[8];
  const float* g1  = (const float*)d_in[9];  const float* b1  = (const float*)d_in[10];
  const float* Wq2 = (const float*)d_in[11]; const float* bq2 = (const float*)d_in[12];
  const float* Wk2 = (const float*)d_in[13]; const float* bk2 = (const float*)d_in[14];
  const float* Wv2 = (const float*)d_in[15]; const float* bv2 = (const float*)d_in[16];
  const float* Wo2 = (const float*)d_in[17]; const float* bo2 = (const float*)d_in[18];
  const float* g2  = (const float*)d_in[19]; const float* b2  = (const float*)d_in[20];
  const float* Wf  = (const float*)d_in[21]; const float* bfb = (const float*)d_in[22];

  char* p = (char*)d_ws;
  auto carve = [&](size_t bytes) -> char* {
    char* r = p; p += (bytes + 255) & ~(size_t)255; return r;
  };
  const size_t EL = (size_t)Mrows * Dm;           // 3,145,728
  __bf16* wtq  = (__bf16*)carve(256 * 768 * 2);
  __bf16* wtk  = (__bf16*)carve(256 * 768 * 2);
  __bf16* wtv  = (__bf16*)carve(256 * 768 * 2);
  __bf16* wtq2 = (__bf16*)carve(384 * 768 * 2);
  __bf16* wtk2 = (__bf16*)carve(384 * 768 * 2);
  __bf16* wtv2 = (__bf16*)carve(768 * 768 * 2);
  __bf16* wto2 = (__bf16*)carve(768 * 768 * 2);
  __bf16* wtf  = (__bf16*)carve(768 * 768 * 2);
  __bf16* encb = (__bf16*)carve(EL * 2);
  float*  imf  = (float*) carve(EL * 4);
  __bf16* imb  = (__bf16*)carve(EL * 2);
  __bf16* Qb   = (__bf16*)carve(EL * 2);
  __bf16* Kb   = (__bf16*)carve(EL * 2);
  __bf16* VtB  = (__bf16*)carve(EL * 2);          // [B,H,64,S]
  float*  aof  = (float*) carve(EL * 4);
  __bf16* aob  = (__bf16*)carve(EL * 2);
  float*  tf   = (float*) carve(EL * 4);
  __bf16* tb   = (__bf16*)carve(EL * 2);
  float*  m2f  = (float*) carve(EL * 4);
  float*  st   = (float*) carve(Dm * 2 * 4);

  const float scale1 = 0.036084391824351615f;     // 1/sqrt(768)
  const float scale2 = 0.125f;                    // 1/sqrt(64)
  const dim3 blk(256);

  // converts (re-run each launch; deterministic)
  k_convT<<<(256 * 768 + 255) / 256, blk, 0, stream>>>(Wq,  wtq,  256, 768);
  k_convT<<<(256 * 768 + 255) / 256, blk, 0, stream>>>(Wk,  wtk,  256, 768);
  k_convT<<<(256 * 768 + 255) / 256, blk, 0, stream>>>(Wv,  wtv,  256, 768);
  k_convT<<<(384 * 768 + 255) / 256, blk, 0, stream>>>(Wq2, wtq2, 384, 768);
  k_convT<<<(384 * 768 + 255) / 256, blk, 0, stream>>>(Wk2, wtk2, 384, 768);
  k_convT<<<(768 * 768 + 255) / 256, blk, 0, stream>>>(Wv2, wtv2, 768, 768);
  k_convT<<<(768 * 768 + 255) / 256, blk, 0, stream>>>(Wo2, wto2, 768, 768);
  k_convT<<<(768 * 768 + 255) / 256, blk, 0, stream>>>(Wf,  wtf,  768, 768);
  k_conv<<<(int)((EL + 255) / 256), blk, 0, stream>>>(encod, encb, (int)EL);
  k_embed<<<(int)((EL + 255) / 256), blk, 0, stream>>>(x, embed, imf, imb);

  const int gemmGrid = 384, flashGrid = 384;
  const int ewGrid = (int)((EL + 255) / 256);

  for (int it = 0; it < 2; ++it) {
    // self attention: Q/K/V projections with ReLU (inputs: thirds of input_multi)
    k_gemm<<<gemmGrid, blk, 0, stream>>>(imb + 0,   Dm, wtq, 256, bq, 1, nullptr, Qb,  nullptr);
    k_gemm<<<gemmGrid, blk, 0, stream>>>(imb + 256, Dm, wtk, 256, bk, 1, nullptr, Kb,  nullptr);
    k_gemm<<<gemmGrid, blk, 0, stream>>>(imb + 512, Dm, wtv, 256, bv, 1, nullptr, nullptr, VtB);
    k_flash<<<flashGrid, blk, 0, stream>>>(Qb, Kb, VtB, scale1, aof, aob);
    // t = BN(input_multi + sa)
    k_bnstats<<<Dm, blk, 0, stream>>>(imf, aof, st);
    k_bnapply<<<ewGrid, blk, 0, stream>>>(imf, aof, st, g1, b1, tf, tb);
    // cross attention: Q2/K2 from encoder halves, V2 from t (no ReLU)
    k_gemm<<<gemmGrid, blk, 0, stream>>>(encb + 0,   Dm, wtq2, 384, bq2, 0, nullptr, Qb, nullptr);
    k_gemm<<<gemmGrid, blk, 0, stream>>>(encb + 384, Dm, wtk2, 384, bk2, 0, nullptr, Kb, nullptr);
    k_gemm<<<gemmGrid, blk, 0, stream>>>(tb,         Dm, wtv2, 768, bv2, 0, nullptr, nullptr, VtB);
    k_flash<<<flashGrid, blk, 0, stream>>>(Qb, Kb, VtB, scale2, aof, aob);
    // m2 = attn_out @ Wo2 + bo2 ; t = BN(m2 + t)
    k_gemm<<<gemmGrid, blk, 0, stream>>>(aob, Dm, wto2, 768, bo2, 0, m2f, nullptr, nullptr);
    k_bnstats<<<Dm, blk, 0, stream>>>(m2f, tf, st);
    k_bnapply<<<ewGrid, blk, 0, stream>>>(m2f, tf, st, g2, b2, tf, tb);
    // ffn = t @ Wf + bf ; input_multi = BN(t + ffn)
    k_gemm<<<gemmGrid, blk, 0, stream>>>(tb, Dm, wtf, 768, bfb, 0, aof, nullptr, nullptr);
    k_bnstats<<<Dm, blk, 0, stream>>>(tf, aof, st);
    k_bnapply<<<ewGrid, blk, 0, stream>>>(tf, aof, st, g2, b2, imf, imb);
  }
  hipMemcpyAsync(d_out, imf, EL * sizeof(float), hipMemcpyDeviceToDevice, stream);
}
